// LSTMAR_84688165142562
// MI455X (gfx1250) — matrix operand order repaired and checked by reading, not compiled, not run on hardware
//
#include <hip/hip_runtime.h>

typedef _Float16 half_t;
typedef _Float16 v16h __attribute__((ext_vector_type(16)));
typedef _Float16 v8h  __attribute__((ext_vector_type(8)));
typedef _Float16 h2   __attribute__((ext_vector_type(2)));
typedef float    v8f  __attribute__((ext_vector_type(8)));

#define HID   20
#define ROWH  72          // staging row stride in halves (144B, 16B aligned, bank-spread)
#define WAVES 8
#define TILEB 16
#define BLK_B (WAVES*TILEB)

// ---------------------------------------------------------------------------
// Prep kernel: swizzle the 4 LSTM layers' weights (+ merged bias as column 31)
// into the WMMA 16-bit A-operand register layout, padded to M=128, K=64.
// blob layout: [layer(4)][operand(16)=mtile*2+khalf][lane(32)][half(16)]
// A-layout used here (16-bit A 16x32, contiguous k, B read the same way): lane%16 = M row; K per half index:
//   K_local = hI + (lane>=16 ? 16 : 0)
// ---------------------------------------------------------------------------
__global__ void prep_weights(
    const float* Wih0, const float* Whh0, const float* b0,
    const float* Wih1, const float* Whh1, const float* b1,
    const float* Wih2, const float* Whh2, const float* b2,
    const float* Wih3, const float* Whh3, const float* b3,
    half_t* blob)
{
    int idx  = blockIdx.x * blockDim.x + threadIdx.x;   // 0..32767
    int hI   = idx & 15;
    int lane = (idx >> 4) & 31;
    int op   = (idx >> 9) & 15;
    int l    = (idx >> 13) & 3;
    int mt = op >> 1, kh = op & 1;
    int M  = mt * 16 + (lane & 15);
    int hi = lane >> 4;
    int Kl = hI + hi * 16;
    int K  = kh * 32 + Kl;

    const float *Wih, *Whh, *b; int in_sz;
    switch (l) {
      case 0:  Wih = Wih0; Whh = Whh0; b = b0; in_sz = 16; break;
      case 1:  Wih = Wih1; Whh = Whh1; b = b1; in_sz = 20; break;
      case 2:  Wih = Wih2; Whh = Whh2; b = b2; in_sz = 4;  break;
      default: Wih = Wih3; Whh = Whh3; b = b3; in_sz = 20; break;
    }
    float v = 0.f;
    int g = M >> 5, r = M & 31;                // gate group (i,f,g,o), row in group
    if (r < HID) {
        int row = g * HID + r;                 // row in the real 80-row matrices
        if (K == 31)                 v = b[row];                       // bias column
        else if (K < 32)             { if (K < in_sz) v = Wih[row * in_sz + K]; }
        else                         { int k2 = K - 32; if (k2 < HID) v = Whh[row * HID + k2]; }
    }
    blob[idx] = (half_t)v;
}

// ---------------------------------------------------------------------------
// Branch-free activations (keep EXEC all-ones through the WMMA pipeline).
// gfx1250 has a hardware v_tanh_f32 transcendental; use it when exposed.
// ---------------------------------------------------------------------------
#if __has_builtin(__builtin_amdgcn_tanhf)
__device__ __forceinline__ float tanh_f(float x) { return __builtin_amdgcn_tanhf(x); }
#else
__device__ __forceinline__ float tanh_f(float x) {
    // 1 - 2/(e^{2x}+1): exp overflow -> inf -> rcp -> 0 -> +1; underflow -> -1.
    float e = __expf(2.f * x);
    return 1.f - 2.f * __builtin_amdgcn_rcpf(e + 1.f);
}
#endif
__device__ __forceinline__ float sigm(float x) {
    return __builtin_fmaf(0.5f, tanh_f(0.5f * x), 0.5f);
}

__device__ __forceinline__ v16h ldsB(const half_t* p) {
    v8h a = *(const v8h*)p;
    v8h b = *(const v8h*)(p + 8);
    return __builtin_shufflevector(a, b, 0,1,2,3,4,5,6,7,8,9,10,11,12,13,14,15);
}

__device__ __forceinline__ float loadEnc(const float* y, const float* x,
                                         const float* u, const float* s,
                                         int b, int t, int k) {
    int bt = b * 8 + t;
    if (k == 0)  return y[bt];
    if (k < 9)   return x[bt * 8 + (k - 1)];
    if (k < 13)  return u[bt * 4 + (k - 9)];
    return s[bt * 3 + (k - 13)];
}

// One LSTM layer-step for a 16-batch column tile:
//   G = W_swizzled x [x|h|1]^T, gate-aligned 8 WMMA-D tiles, then c/h update.
// dst0/dst1 already include the destination k-slot offset.
__device__ __forceinline__ void layer_step(int L, const half_t* blob,
                                           const half_t* bufIn,
                                           half_t* dst0, half_t* dst1,
                                           float c[2][8], int lane)
{
    int n = lane & 15, hi = lane >> 4;
    const half_t* brow = bufIn + n * ROWH + hi * 16;
    v16h B0 = ldsB(brow);          // K = 0..31  of [x|h|1]
    v16h B1 = ldsB(brow + 32);     // K = 32..63
    #pragma unroll
    for (int t = 0; t < 2; ++t) {
        v8f G[4];
        #pragma unroll
        for (int g = 0; g < 4; ++g) {
            int mt = g * 2 + t;
            const half_t* a0 = blob + (size_t)((L * 16 + mt * 2 + 0) * 512) + lane * 16;
            const half_t* a1 = blob + (size_t)((L * 16 + mt * 2 + 1) * 512) + lane * 16;
            v16h A0 = *(const v16h*)a0;
            v16h A1 = *(const v16h*)a1;
            v8f acc = {};
            acc = __builtin_amdgcn_wmma_f32_16x16x32_f16(false, A0, false, B0,
                                                         (short)0, acc, false, false);
            acc = __builtin_amdgcn_wmma_f32_16x16x32_f16(false, A1, false, B1,
                                                         (short)0, acc, false, false);
            G[g] = acc;
        }
        float h[8];
        #pragma unroll
        for (int j = 0; j < 8; ++j) {
            float i_ = sigm(G[0][j]);
            float f_ = sigm(G[1][j]);
            float g_ = tanh_f(G[2][j]);
            float o_ = sigm(G[3][j]);
            float cn = f_ * c[t][j] + i_ * g_;
            c[t][j] = cn;
            h[j] = o_ * tanh_f(cn);
        }
        // store only real rows (m < 20); padded rows stay exactly zero in LDS
        int mbase = 16 * t + 8 * hi;
        #pragma unroll
        for (int j = 0; j < 8; j += 2) {
            if (t == 0 || (j < 4 && hi == 0)) {     // m = mbase+j < 20 only
                h2 p; p.x = (half_t)h[j]; p.y = (half_t)h[j + 1];
                *(h2*)(dst0 + n * ROWH + (mbase + j)) = p;
                if (dst1) *(h2*)(dst1 + n * ROWH + (mbase + j)) = p;
            }
        }
    }
}

__global__ void __launch_bounds__(256)
lstm_main(const float* y_past, const float* x_past, const float* u_past,
          const float* s_past, const float* u_future, const half_t* blob,
          const float* fW1, const float* fb1, const float* fW2, const float* fb2,
          const float* fW3, const float* fb3, float* out)
{
    __shared__ __align__(16) half_t stage[WAVES][2][16 * ROWH];
    int tid = threadIdx.x;
    int w = tid >> 5, lane = tid & 31;
    int n = lane & 15, hi = lane >> 4;
    int b = blockIdx.x * BLK_B + w * TILEB + n;

    half_t* st0 = &stage[w][0][0];   // layer0 input buffer: [x | h0_state | 1]
    half_t* st1 = &stage[w][1][0];   // layer1 input buffer: [h0 | h1_state | 1]

    // zero staging, plant the constant-1 bias activation at k=31 of each slot0
    {
        uint32_t* p = (uint32_t*)st0;
        for (int i = lane; i < (2 * 16 * ROWH) / 2; i += 32) p[i] = 0u;
    }
    __syncthreads();
    if (hi == 0) {
        st0[n * ROWH + 31] = (half_t)1.f;
        st1[n * ROWH + 31] = (half_t)1.f;
    }
    __syncthreads();

    float c0[2][8], c1[2][8];
    #pragma unroll
    for (int t = 0; t < 2; ++t)
        #pragma unroll
        for (int j = 0; j < 8; ++j) { c0[t][j] = 0.f; c1[t][j] = 0.f; }

    // ---------------- encoder: 8 steps over concat(y,x,u,s) ----------------
    for (int t = 0; t < 8; ++t) {
        #pragma unroll
        for (int j = 0; j < 8; j += 2) {        // lanes cover k = 0..15
            int k0 = hi * 8 + j;
            float f0 = loadEnc(y_past, x_past, u_past, s_past, b, t, k0);
            float f1 = loadEnc(y_past, x_past, u_past, s_past, b, t, k0 + 1);
            h2 p; p.x = (half_t)f0; p.y = (half_t)f1;
            *(h2*)(st0 + n * ROWH + k0) = p;
        }
        __syncthreads();
        layer_step(0, blob, st0, st1,      st0 + 32, c0, lane);  // h0 -> l1 input + recur
        __syncthreads();
        layer_step(1, blob, st1, st1 + 32, nullptr,  c1, lane);  // h1 -> recur
        __syncthreads();
    }

    // ---------------- decoder: 4 steps over u_future (state carried) -------
    if (hi == 0) {                               // clear stale encoder x, k=4..15
        #pragma unroll
        for (int k = 4; k < 16; k += 2) {
            h2 z; z.x = (half_t)0.f; z.y = (half_t)0.f;
            *(h2*)(st0 + n * ROWH + k) = z;
        }
    }
    __syncthreads();

    for (int t = 0; t < 4; ++t) {
        if (hi == 0) {
            int bt = b * 4 + t;
            #pragma unroll
            for (int k = 0; k < 4; k += 2) {
                h2 p; p.x = (half_t)u_future[bt * 4 + k];
                      p.y = (half_t)u_future[bt * 4 + k + 1];
                *(h2*)(st0 + n * ROWH + k) = p;
            }
        }
        __syncthreads();
        layer_step(2, blob, st0, st1,      st0 + 32, c0, lane);
        __syncthreads();
        layer_step(3, blob, st1, st1 + 32, nullptr,  c1, lane);
        __syncthreads();

        // FCNN head 20 -> 10 -> 10 -> 1 on the decoder layer-1 output
        if (hi == 0) {
            float hb[20];
            #pragma unroll
            for (int k = 0; k < 20; ++k) hb[k] = (float)st1[n * ROWH + 32 + k];
            float z1[10];
            #pragma unroll
            for (int j = 0; j < 10; ++j) {
                float a = fb1[j];
                #pragma unroll
                for (int k = 0; k < 20; ++k) a += fW1[j * 20 + k] * hb[k];
                z1[j] = a > 0.f ? a : 0.f;
            }
            float z2[10];
            #pragma unroll
            for (int j = 0; j < 10; ++j) {
                float a = fb2[j];
                #pragma unroll
                for (int k = 0; k < 10; ++k) a += fW2[j * 10 + k] * z1[k];
                z2[j] = a > 0.f ? a : 0.f;
            }
            float a = fb3[0];
            #pragma unroll
            for (int k = 0; k < 10; ++k) a += fW3[k] * z2[k];
            out[b * 4 + t] = a;
        }
        __syncthreads();
    }
}

// ---------------------------------------------------------------------------
extern "C" void kernel_launch(void* const* d_in, const int* in_sizes, int n_in,
                              void* d_out, int out_size, void* d_ws, size_t ws_size,
                              hipStream_t stream)
{
    const float* y_past   = (const float*)d_in[0];
    const float* x_past   = (const float*)d_in[1];
    const float* u_past   = (const float*)d_in[2];
    const float* s_past   = (const float*)d_in[3];
    const float* u_future = (const float*)d_in[4];
    const float* Wih1_0 = (const float*)d_in[5];
    const float* Whh1_0 = (const float*)d_in[6];
    const float* b1_0   = (const float*)d_in[7];
    const float* Wih1_1 = (const float*)d_in[8];
    const float* Whh1_1 = (const float*)d_in[9];
    const float* b1_1   = (const float*)d_in[10];
    const float* Wih2_0 = (const float*)d_in[11];
    const float* Whh2_0 = (const float*)d_in[12];
    const float* b2_0   = (const float*)d_in[13];
    const float* Wih2_1 = (const float*)d_in[14];
    const float* Whh2_1 = (const float*)d_in[15];
    const float* b2_1   = (const float*)d_in[16];
    const float* fW1 = (const float*)d_in[17];
    const float* fb1 = (const float*)d_in[18];
    const float* fW2 = (const float*)d_in[19];
    const float* fb2 = (const float*)d_in[20];
    const float* fW3 = (const float*)d_in[21];
    const float* fb3 = (const float*)d_in[22];

    half_t* blob = (half_t*)d_ws;                    // 64 KB swizzled weight blob
    prep_weights<<<128, 256, 0, stream>>>(Wih1_0, Whh1_0, b1_0,
                                          Wih1_1, Whh1_1, b1_1,
                                          Wih2_0, Whh2_0, b2_0,
                                          Wih2_1, Whh2_1, b2_1, blob);

    int B = in_sizes[0] / 8;                          // y_past is [B,8,1]
    lstm_main<<<B / BLK_B, 256, 0, stream>>>(y_past, x_past, u_past, s_past,
                                             u_future, blob,
                                             fW1, fb1, fW2, fb2, fW3, fb3,
                                             (float*)d_out);
    (void)n_in; (void)out_size; (void)ws_size;
}
